// HyperGraphModel_12275016532626
// MI455X (gfx1250) — compile-verified, matrix operand order checked by reading, not run on hardware
//
#include <hip/hip_runtime.h>

// ---------------------------------------------------------------------------
// HyperGraph conv model on MI455X (gfx1250), bf16 WMMA with fp32 accumulate.
// Heavy ops are the four H-GEMMs over the 640MB dense incidence matrix:
// HBM-bound (~2.56GB -> ~110us at 23.3TB/s). The H stream now uses CDNA5
// GLOBAL_LOAD_ASYNC_TO_LDS_B128 (ASYNCcnt) directly into LDS (fp32), with
// bf16 conversion folded into the WMMA fragment build.
// ---------------------------------------------------------------------------

typedef __attribute__((ext_vector_type(16))) __bf16 v16bf;
typedef __attribute__((ext_vector_type(8)))  float  v8f;

#define Nn 20000
#define Ee 8000
#define Dd 128
#define Tt 5

#define BM 64          // block tile rows
#define BK 32          // K step (matches WMMA K=32)
#define THREADS 128    // 4 waves of 32
#define LDA 33         // LDS row stride (bf16 elems) for bf16 A tile [BM][LDA]
#define LDB 33         // LDS row stride for B-transposed tile [128][LDB]
#define LDAF 36        // fp32 A tile row stride (floats), 144B (16B aligned)
#define LDEF 68        // fp32 A^T tile row stride (floats), 272B (16B aligned)

// ---- CDNA5 async global->LDS helpers --------------------------------------
__device__ __forceinline__ unsigned lds_off(const void* p) {
    // Flat addresses in the LDS aperture carry the LDS byte address in the
    // low 32 bits (ISA 10.2).
    return (unsigned)(uintptr_t)p;
}
__device__ __forceinline__ void async_ld_b128(unsigned lds, const float* g) {
    asm volatile("global_load_async_to_lds_b128 %0, %1, off"
                 :: "v"(lds), "v"(g) : "memory");
}
__device__ __forceinline__ void wait_async0() {
    asm volatile("s_wait_asynccnt 0x0" ::: "memory");
}

__device__ __forceinline__ int kbase(int v, int hi) {
    // 16-bit A/B fragment K placement (ISA 7.12.2): VGPR v<4 -> K=2v, v>=4 -> K=8+2v,
    // lanes 16-31 shifted by +8.
    return 2 * v + (v >= 4 ? 8 : 0) + hi * 8;
}

__device__ __forceinline__ v16bf load_a_frag(const __bf16* As, int lane, int w) {
    int ar = 16 * w + (lane & 15);
    int hi = lane >> 4;
    v16bf a;
#pragma unroll
    for (int v = 0; v < 8; ++v) {
        int kb = kbase(v, hi);
        a[2 * v]     = As[ar * LDA + kb];
        a[2 * v + 1] = As[ar * LDA + kb + 1];
    }
    return a;
}

// A fragment from fp32 row-major LDS tile [BM][LDAF], cvt to bf16 on the fly.
__device__ __forceinline__ v16bf load_a_frag_f32(const float* As32, int lane, int w) {
    int ar = 16 * w + (lane & 15);
    int hi = lane >> 4;
    v16bf a;
#pragma unroll
    for (int v = 0; v < 8; ++v) {
        int kb = kbase(v, hi);
        float2 p = *(const float2*)&As32[ar * LDAF + kb];
        a[2 * v]     = (__bf16)p.x;
        a[2 * v + 1] = (__bf16)p.y;
    }
    return a;
}

// A fragment from fp32 K-major LDS tile [BK][LDEF] (A^T staging for H^T GEMM).
__device__ __forceinline__ v16bf load_a_frag_f32t(const float* Ak32, int lane, int w) {
    int ar = 16 * w + (lane & 15);
    int hi = lane >> 4;
    v16bf a;
#pragma unroll
    for (int v = 0; v < 8; ++v) {
        int kb = kbase(v, hi);
        a[2 * v]     = (__bf16)Ak32[kb * LDEF + ar];
        a[2 * v + 1] = (__bf16)Ak32[(kb + 1) * LDEF + ar];
    }
    return a;
}

__device__ __forceinline__ v16bf load_b_frag(const __bf16* Bs, int lane, int jt) {
    int n  = jt * 16 + (lane & 15);
    int hi = lane >> 4;
    v16bf b;
#pragma unroll
    for (int v = 0; v < 8; ++v) {
        int kb = kbase(v, hi);
        b[2 * v]     = Bs[n * LDB + kb];
        b[2 * v + 1] = Bs[n * LDB + kb + 1];
    }
    return b;
}

// ---------------------------------------------------------------------------
// proj: X0[n,:] = emb[n,:] @ Wt[type_ids[n]] + bt[type_ids[n]]
// Implemented as sum_t mask_t(emb) @ Wt[t]; mask is uniform per lane-row.
// ---------------------------------------------------------------------------
__global__ __launch_bounds__(THREADS)
void proj_kernel(const float* __restrict__ emb, const float* __restrict__ Wt,
                 const float* __restrict__ bt, const int* __restrict__ tids,
                 float* __restrict__ X0) {
    __shared__ __bf16 As[BM * LDA];
    __shared__ __bf16 Bs[Tt * Dd * LDB];   // 5 type matrices per K-step

    const int tid = threadIdx.x, lane = tid & 31, w = tid >> 5;
    const int row0 = blockIdx.x * BM;

    const int arow   = row0 + 16 * w + (lane & 15);
    const int mytype = (arow < Nn) ? tids[arow] : -1;

    v8f acc[8];
    {
        v8f z = {};
#pragma unroll
        for (int j = 0; j < 8; ++j) acc[j] = z;
    }

    for (int k0 = 0; k0 < Dd; k0 += BK) {
#pragma unroll
        for (int i = 0; i < (BM * BK) / THREADS; ++i) {
            int idx = tid + i * THREADS;
            int r = idx >> 5, c = idx & 31;
            int gr = row0 + r;
            float v = (gr < Nn) ? emb[(size_t)gr * Dd + k0 + c] : 0.f;
            As[r * LDA + c] = (__bf16)v;
        }
#pragma unroll
        for (int i = 0; i < (Tt * BK * Dd) / THREADS; ++i) {
            int idx = tid + i * THREADS;
            int t   = idx >> 12;
            int rem = idx & 4095;
            int r = rem >> 7, c = rem & 127;
            Bs[t * Dd * LDB + c * LDB + r] =
                (__bf16)Wt[(size_t)t * Dd * Dd + (size_t)(k0 + r) * Dd + c];
        }
        __syncthreads();

        v16bf a = load_a_frag(As, lane, w);
        v16bf z = {};
#pragma unroll
        for (int t = 0; t < Tt; ++t) {
            v16bf at = (mytype == t) ? a : z;
#pragma unroll
            for (int jt = 0; jt < 8; ++jt) {
                v16bf bf = load_b_frag(Bs + t * Dd * LDB, lane, jt);
                acc[jt] = __builtin_amdgcn_wmma_f32_16x16x32_bf16(
                    false, at, false, bf, (short)0, acc[jt], false, false);
            }
        }
        __syncthreads();
    }

    const int hi = lane >> 4, nl = lane & 15;
#pragma unroll
    for (int jt = 0; jt < 8; ++jt) {
        int col = jt * 16 + nl;
#pragma unroll
        for (int v = 0; v < 8; ++v) {
            int grow = row0 + 16 * w + v + 8 * hi;
            if (grow < Nn) {
                int t = tids[grow];
                X0[(size_t)grow * Dd + col] = acc[jt][v] + bt[t * Dd + col];
            }
        }
    }
}

// ---------------------------------------------------------------------------
// xw: Y[n,:] = dv[n] * (X[n,:] @ W + b)      (dv folded in so H^T consumes it)
// ---------------------------------------------------------------------------
__global__ __launch_bounds__(THREADS)
void xw_kernel(const float* __restrict__ X, const float* __restrict__ W,
               const float* __restrict__ b, const float* __restrict__ dv,
               float* __restrict__ Y) {
    __shared__ __bf16 As[BM * LDA];
    __shared__ __bf16 Bs[Dd * LDB];

    const int tid = threadIdx.x, lane = tid & 31, w = tid >> 5;
    const int row0 = blockIdx.x * BM;

    v8f acc[8];
    {
        v8f z = {};
#pragma unroll
        for (int j = 0; j < 8; ++j) acc[j] = z;
    }

    for (int k0 = 0; k0 < Dd; k0 += BK) {
#pragma unroll
        for (int i = 0; i < (BM * BK) / THREADS; ++i) {
            int idx = tid + i * THREADS;
            int r = idx >> 5, c = idx & 31;
            int gr = row0 + r;
            float v = (gr < Nn) ? X[(size_t)gr * Dd + k0 + c] : 0.f;
            As[r * LDA + c] = (__bf16)v;
        }
#pragma unroll
        for (int i = 0; i < (BK * Dd) / THREADS; ++i) {
            int idx = tid + i * THREADS;
            int r = idx >> 7, c = idx & 127;
            Bs[c * LDB + r] = (__bf16)W[(size_t)(k0 + r) * Dd + c];
        }
        __syncthreads();

        v16bf a = load_a_frag(As, lane, w);
#pragma unroll
        for (int jt = 0; jt < 8; ++jt) {
            v16bf bf = load_b_frag(Bs, lane, jt);
            acc[jt] = __builtin_amdgcn_wmma_f32_16x16x32_bf16(
                false, a, false, bf, (short)0, acc[jt], false, false);
        }
        __syncthreads();
    }

    const int hi = lane >> 4, nl = lane & 15;
#pragma unroll
    for (int jt = 0; jt < 8; ++jt) {
        int col = jt * 16 + nl;
#pragma unroll
        for (int v = 0; v < 8; ++v) {
            int grow = row0 + 16 * w + v + 8 * hi;
            if (grow < Nn)
                Y[(size_t)grow * Dd + col] = (acc[jt][v] + b[col]) * dv[grow];
        }
    }
}

// ---------------------------------------------------------------------------
// ht: m[e,:] = de[e] * sum_n H[n,e] * Xw[n,:]   (A = H^T)
// A tile staged K-major in fp32 LDS via GLOBAL_LOAD_ASYNC_TO_LDS_B128:
// As32k[k][e] <- H[(k0+k)*Ee + e0 + e]  (contiguous along e).
// M = E (8000, /64 exact), K = N (20000, /32 exact) -> no guards.
// ---------------------------------------------------------------------------
__global__ __launch_bounds__(THREADS)
void ht_kernel(const float* __restrict__ H, const float* __restrict__ Xw,
               const float* __restrict__ de, float* __restrict__ Mo) {
    __shared__ float  As32k[BK * LDEF];    // 8.5 KB fp32, K-major
    __shared__ __bf16 Bs[Dd * LDB];

    const int tid = threadIdx.x, lane = tid & 31, w = tid >> 5;
    const int e0 = blockIdx.x * BM;

    v8f acc[8];
    {
        v8f z = {};
#pragma unroll
        for (int j = 0; j < 8; ++j) acc[j] = z;
    }

    for (int k0 = 0; k0 < Nn; k0 += BK) {
        // async A tile: 512 x b128 chunks, 4 per thread
#pragma unroll
        for (int i = 0; i < 4; ++i) {
            int q  = tid + i * THREADS;    // 0..511
            int c  = q >> 4;               // k local 0..31
            int r4 = (q & 15) << 2;        // e offset 0..60
            async_ld_b128(lds_off(&As32k[c * LDEF + r4]),
                          &H[(size_t)(k0 + c) * Ee + e0 + r4]);
        }
        // B tile (Xw rows, L2-resident) via VGPR path, bf16 transposed
#pragma unroll
        for (int i = 0; i < (BK * Dd) / THREADS; ++i) {
            int idx = tid + i * THREADS;
            int r = idx >> 7, c = idx & 127;
            Bs[c * LDB + r] = (__bf16)Xw[(size_t)(k0 + r) * Dd + c];
        }
        wait_async0();
        __syncthreads();

        v16bf a = load_a_frag_f32t(As32k, lane, w);
#pragma unroll
        for (int jt = 0; jt < 8; ++jt) {
            v16bf bf = load_b_frag(Bs, lane, jt);
            acc[jt] = __builtin_amdgcn_wmma_f32_16x16x32_bf16(
                false, a, false, bf, (short)0, acc[jt], false, false);
        }
        __syncthreads();
    }

    const int hi = lane >> 4, nl = lane & 15;
#pragma unroll
    for (int jt = 0; jt < 8; ++jt) {
        int col = jt * 16 + nl;
#pragma unroll
        for (int v = 0; v < 8; ++v) {
            int ge = e0 + 16 * w + v + 8 * hi;
            Mo[(size_t)ge * Dd + col] = acc[jt][v] * de[ge];
        }
    }
}

// ---------------------------------------------------------------------------
// h: Y[n,:] = dv[n] * sum_e H[n,e] * m[e,:]
// A tile (rows of H) staged fp32 row-major in LDS via async-to-LDS b128.
// FUSE=true additionally emits out = w0*X0 + w1*X1 + w2*Y (softmax fusion).
// ---------------------------------------------------------------------------
template <bool FUSE>
__global__ __launch_bounds__(THREADS)
void h_kernel(const float* __restrict__ H, const float* __restrict__ Mi,
              const float* __restrict__ dv, float* __restrict__ Y,
              const float* __restrict__ X0, const float* __restrict__ X1,
              const float* __restrict__ fw) {
    __shared__ float  As32[BM * LDAF];     // 9.2 KB fp32, row-major
    __shared__ __bf16 Bs[Dd * LDB];

    const int tid = threadIdx.x, lane = tid & 31, w = tid >> 5;
    const int row0 = blockIdx.x * BM;

    float w0 = 0.f, w1 = 0.f, w2 = 1.f;
    if (FUSE) {
        float f0 = fw[0], f1 = fw[1], f2 = fw[2];
        float mx = fmaxf(f0, fmaxf(f1, f2));
        float e0 = __expf(f0 - mx), e1 = __expf(f1 - mx), e2 = __expf(f2 - mx);
        float inv = 1.f / (e0 + e1 + e2);
        w0 = e0 * inv; w1 = e1 * inv; w2 = e2 * inv;
    }

    // Boundary block: zero the tile once; OOB rows never get async-overwritten.
    if (row0 + BM > Nn) {
        for (int i = tid; i < BM * LDAF; i += THREADS) As32[i] = 0.f;
        __syncthreads();
    }

    v8f acc[8];
    {
        v8f z = {};
#pragma unroll
        for (int j = 0; j < 8; ++j) acc[j] = z;
    }

    for (int k0 = 0; k0 < Ee; k0 += BK) {
        // async A tile: As32[r][k] <- H[(row0+r)*Ee + k0 + k], 4x b128/thread
#pragma unroll
        for (int i = 0; i < 4; ++i) {
            int q  = tid + i * THREADS;    // 0..511
            int r  = q >> 3;               // row local 0..63
            int c4 = (q & 7) << 2;         // k offset 0..28
            int gr = row0 + r;
            if (gr < Nn)
                async_ld_b128(lds_off(&As32[r * LDAF + c4]),
                              &H[(size_t)gr * Ee + k0 + c4]);
        }
        // B tile (m rows, L2-resident) via VGPR path, bf16 transposed
#pragma unroll
        for (int i = 0; i < (BK * Dd) / THREADS; ++i) {
            int idx = tid + i * THREADS;
            int r = idx >> 7, c = idx & 127;
            Bs[c * LDB + r] = (__bf16)Mi[(size_t)(k0 + r) * Dd + c];
        }
        wait_async0();
        __syncthreads();

        v16bf a = load_a_frag_f32(As32, lane, w);
#pragma unroll
        for (int jt = 0; jt < 8; ++jt) {
            v16bf bf = load_b_frag(Bs, lane, jt);
            acc[jt] = __builtin_amdgcn_wmma_f32_16x16x32_bf16(
                false, a, false, bf, (short)0, acc[jt], false, false);
        }
        __syncthreads();
    }

    const int hi = lane >> 4, nl = lane & 15;
#pragma unroll
    for (int jt = 0; jt < 8; ++jt) {
        int col = jt * 16 + nl;
#pragma unroll
        for (int v = 0; v < 8; ++v) {
            int grow = row0 + 16 * w + v + 8 * hi;
            if (grow < Nn) {
                size_t idx = (size_t)grow * Dd + col;
                float x2 = acc[jt][v] * dv[grow];
                if (FUSE)
                    Y[idx] = w0 * X0[idx] + w1 * X1[idx] + w2 * x2;
                else
                    Y[idx] = x2;
            }
        }
    }
}

// ---------------------------------------------------------------------------
extern "C" void kernel_launch(void* const* d_in, const int* in_sizes, int n_in,
                              void* d_out, int out_size, void* d_ws, size_t ws_size,
                              hipStream_t stream) {
    const float* emb = (const float*)d_in[0];
    const float* Wt  = (const float*)d_in[1];
    const float* bt  = (const float*)d_in[2];
    const float* W1  = (const float*)d_in[3];
    const float* b1  = (const float*)d_in[4];
    const float* W2  = (const float*)d_in[5];
    const float* b2  = (const float*)d_in[6];
    const float* fw  = (const float*)d_in[7];
    const float* H   = (const float*)d_in[8];
    const float* dv  = (const float*)d_in[9];
    const float* de  = (const float*)d_in[10];
    const int*   tid = (const int*)d_in[11];
    float* out = (float*)d_out;

    // workspace: X0, X1, Xw (N*D each) + m (E*D)  => ~34.8 MB fp32
    float* X0 = (float*)d_ws;
    float* X1 = X0 + (size_t)Nn * Dd;
    float* Xw = X1 + (size_t)Nn * Dd;
    float* Mb = Xw + (size_t)Nn * Dd;

    dim3 blk(THREADS);
    const int gridN = (Nn + BM - 1) / BM;   // 313
    const int gridE = Ee / BM;              // 125

    proj_kernel<<<gridN, blk, 0, stream>>>(emb, Wt, bt, tid, X0);

    xw_kernel<<<gridN, blk, 0, stream>>>(X0, W1, b1, dv, Xw);
    ht_kernel<<<gridE, blk, 0, stream>>>(H, Xw, de, Mb);
    h_kernel<false><<<gridN, blk, 0, stream>>>(H, Mb, dv, X1, nullptr, nullptr, nullptr);

    xw_kernel<<<gridN, blk, 0, stream>>>(X1, W2, b2, dv, Xw);
    ht_kernel<<<gridE, blk, 0, stream>>>(H, Xw, de, Mb);
    h_kernel<true><<<gridN, blk, 0, stream>>>(H, Mb, dv, out, X0, X1, fw);
}
